// FDKN_58634893525162
// MI455X (gfx1250) — matrix operand order repaired and checked by reading, not compiled, not run on hardware
//
#include <hip/hip_runtime.h>
#include <math.h>

// ---------------------------------------------------------------------------
// FDKN depth super-resolution on gfx1250 (MI455X).
// Convs = implicit-GEMM on V_WMMA_F32_16X16X32_BF16 (wave32), BN folded,
// bf16 activations.  Weights pre-permuted to fragment order so each K-panel
// is DMA'd straight into LDS with GLOBAL_LOAD_ASYNC_TO_LDS_B128 (ASYNCcnt).
// A-tile staging exploits block==image-row so per-element address math is
// hoisted to one division per K-step per thread.
// ---------------------------------------------------------------------------

typedef __bf16 v16bf __attribute__((ext_vector_type(16)));
typedef float  v8f   __attribute__((ext_vector_type(8)));

#define HW_LR   16384      // 128*128
#define W_LR    128
#define H_LR    128
#define HW_HR   262144     // 512*512
#define W_HR    512

__device__ __forceinline__ unsigned short f2bf(float f) {
    unsigned int u = __builtin_bit_cast(unsigned int, f);
    unsigned int r = u + 0x7fffu + ((u >> 16) & 1u);   // round-to-nearest-even
    return (unsigned short)(r >> 16);
}
__device__ __forceinline__ float bf2f(unsigned short h) {
    unsigned int u = ((unsigned int)h) << 16;
    return __builtin_bit_cast(float, u);
}
__device__ __forceinline__ int iclamp(int v, int lo, int hi) {
    return v < lo ? lo : (v > hi ? hi : v);
}

// ---------------------------------------------------------------------------
// Bicubic 4x upsample (Keys a=-0.5, half-pixel centers, edge clamp)
// ---------------------------------------------------------------------------
__device__ __forceinline__ float cub_w(float t) {
    const float a = -0.5f;
    float at = fabsf(t);
    if (at <= 1.f) return ((a + 2.f) * at - (a + 3.f)) * at * at + 1.f;
    if (at <  2.f) return (((at - 5.f) * at + 8.f) * at - 4.f) * a;
    return 0.f;
}

__global__ __launch_bounds__(256) void bicubic_up(const float* __restrict__ x,
                                                  float* __restrict__ xu) {
    int idx = blockIdx.x * 256 + threadIdx.x;
    if (idx >= HW_HR) return;
    int oy = idx >> 9, ox = idx & 511;
    float fy = (oy + 0.5f) * 0.25f - 0.5f;
    float fx = (ox + 0.5f) * 0.25f - 0.5f;
    int y0 = (int)floorf(fy), x0 = (int)floorf(fx);
    float wy[4], wx[4];
#pragma unroll
    for (int i = 0; i < 4; ++i) {
        wy[i] = cub_w(fy - (float)(y0 - 1 + i));
        wx[i] = cub_w(fx - (float)(x0 - 1 + i));
    }
    float acc = 0.f;
#pragma unroll
    for (int i = 0; i < 4; ++i) {
        int yy = iclamp(y0 - 1 + i, 0, H_LR - 1);
        float row = 0.f;
#pragma unroll
        for (int j = 0; j < 4; ++j) {
            int xx = iclamp(x0 - 1 + j, 0, W_LR - 1);
            row += wx[j] * x[yy * W_LR + xx];
        }
        acc += wy[i] * row;
    }
    xu[idx] = acc;
}

// ---------------------------------------------------------------------------
// Space-to-depth (s=4), f32 -> bf16.  out[(i*4+j)*C + c][y][x] = in[c][4y+i][4x+j]
// ---------------------------------------------------------------------------
__global__ __launch_bounds__(256) void s2d_bf16(const float* __restrict__ in,
                                                unsigned short* __restrict__ out,
                                                int C) {
    int idx = blockIdx.x * 256 + threadIdx.x;
    int total = C * 16 * HW_LR;
    if (idx >= total) return;
    int hw = idx & (HW_LR - 1);
    int ch = idx >> 14;
    int y = hw >> 7, x = hw & 127;
    int ij = ch / C, c = ch - ij * C;
    int i = ij >> 2, j = ij & 3;
    out[idx] = f2bf(in[(c * W_HR + (y * 4 + i)) * W_HR + (x * 4 + j)]);
}

// ---------------------------------------------------------------------------
// Weight prep: OIHW f32 -> fragment-ordered bf16 panels with BN folded in.
// Panel p = k/32 holds COUTp/16 subtiles; each subtile is the exact 1KB LDS
// image [lane(32)][e(16)] consumed by the WMMA B fragment:
//   lane = ((k>>4)&1)*16 + (n&15),  e = k&15.
// k >= K (K padded to mult of 32) and o >= COUTr are zero-filled.
// ---------------------------------------------------------------------------
__global__ __launch_bounds__(256) void prep_weights(
    const float* __restrict__ w, const float* __restrict__ b,
    const float* __restrict__ g, const float* __restrict__ bb,
    const float* __restrict__ m, const float* __restrict__ v,
    int K, int Kp, int COUTr, int COUTp,
    unsigned short* __restrict__ wout, float* __restrict__ bout) {
    int idx = blockIdx.x * 256 + threadIdx.x;
    int total = Kp * COUTp;
    if (idx < total) {
        int k = idx / COUTp, o = idx - k * COUTp;
        float val = 0.f;
        if (k < K && o < COUTr) {
            float s = 1.f;
            if (g) s = g[o] * rsqrtf(v[o] + 1e-5f);
            val = w[o * K + k] * s;          // OIHW flat: o*K + (i*ks*ks + kh*ks + kw)
        }
        int dst = ((k >> 5) * (COUTp >> 4) + (o >> 4)) * 512
                + (((k >> 4) & 1) * 16 + (o & 15)) * 16 + (k & 15);
        wout[dst] = f2bf(val);
    }
    if (idx < COUTp && bout) {
        float bias = 0.f;
        if (idx < COUTr) {
            bias = b[idx];
            if (g) bias = (bias - m[idx]) * (g[idx] * rsqrtf(v[idx] + 1e-5f)) + bb[idx];
        }
        bout[idx] = bias;
    }
}

// ---------------------------------------------------------------------------
// Implicit-GEMM conv via WMMA bf16.
//   M = 16384 pixels, N = COUT (padded), K = CIN*KS*KS.
//   Block: 256 threads = 8 waves (4 M-groups x 2 N-groups).
//   Block tile: M=128 (exactly one LR image row), N=32*NT.
//   Each wave: 2 M-subtiles x NT N-subtiles -> 2*NT WMMAs per K-step.
//   B panel: async DMA (GLOBAL_LOAD_ASYNC_TO_LDS_B128) of the pre-permuted
//   contiguous panel; A tile: row-gather with hoisted address math.
//   ACT: 0 = linear (f32 out), 1 = relu (bf16 out), 2 = sigmoid (f32 out).
// ---------------------------------------------------------------------------
template <int CIN, int KS, int COUT, int ACT, int NT>
__global__ __launch_bounds__(256) void conv_wmma(
    const unsigned short* __restrict__ act_in,   // [CIN][HW_LR] bf16
    const unsigned short* __restrict__ wgt,      // fragment-ordered panels
    const float* __restrict__ bias,              // [COUT]
    unsigned short* __restrict__ out_bf,         // [COUT][HW_LR] (ACT==1)
    float* __restrict__ out_f32)                 // [COUT][HW_LR] (ACT!=1)
{
    constexpr int K = CIN * KS * KS;

    __shared__ __align__(32) unsigned short Asf[8][32][16];      // 8 M-subtiles
    __shared__ __align__(32) unsigned short Bsf[2 * NT][32][16]; // 2*NT N-subtiles

    const int tid  = threadIdx.x;
    const int lane = tid & 31;
    const int wave = tid >> 5;
    const int mg   = wave & 3;     // M group (2 subtiles each)
    const int ng   = wave >> 2;    // N group (NT subtiles each)
    const int m0   = blockIdx.x * 128;
    const int n0   = blockIdx.y * (32 * NT);
    const int hi   = lane >> 4;
    const int l16  = lane & 15;

    // A staging role (fixed per thread): this thread stages K-column kl of
    // pixels mlb, mlb+8, ... (block == one LR image row y = blockIdx.x).
    const int kl   = tid & 31;
    const int mlb  = tid >> 5;
    const int ahi  = kl >> 4;
    const int ae   = kl & 15;
    const int yrow = blockIdx.x;

    typedef unsigned short __attribute__((address_space(3))) lds_u16;
    const unsigned int bs_lds = (unsigned int)(size_t)(lds_u16*)&Bsf[0][0][0];

    v8f acc[2][NT];
#pragma unroll
    for (int i = 0; i < 2; ++i)
#pragma unroll
        for (int j = 0; j < NT; ++j)
            acc[i][j] = (v8f){0.f, 0.f, 0.f, 0.f, 0.f, 0.f, 0.f, 0.f};

    for (int kk0 = 0; kk0 < K; kk0 += 32) {
        // ---- async DMA: fragment-ordered B panel -> LDS (overlaps A gather) ----
        {
            const unsigned short* wpan =
                wgt + ((size_t)(kk0 >> 5) * (COUT >> 4) + (n0 >> 4)) * 512;
            if (tid < 128 * NT) {
                unsigned int goff = (unsigned int)tid * 16u;
                asm volatile(
                    "global_load_async_to_lds_b128 %0, %1, %2 offset:0"
                    :: "v"(bs_lds + goff), "v"(goff), "s"(wpan)
                    : "memory");
            }
        }
        // ---- A tile: one (ci,kh,kw) column of this image row, permuted ----
        int kk = kk0 + kl;
        bool rowok = false;
        int rowbase = 0, dx = 0;
        if (kk < K) {
            int ci, kh, kw;
            if (KS == 1) { ci = kk; kh = 1; kw = 1; }
            else {
                ci = kk / 9;
                int rem = kk - ci * 9;
                kh = rem / 3;
                kw = rem - kh * 3;
            }
            int iy = yrow + kh - 1;
            rowok = (iy >= 0) && (iy < H_LR);
            rowbase = ci * HW_LR + iy * W_LR;
            dx = kw - 1;
        }
#pragma unroll
        for (int it = 0; it < 16; ++it) {
            int ml = mlb + (it << 3);
            int ix = ml + dx;
            unsigned short v = 0;
            if (rowok && (unsigned)ix < (unsigned)W_LR) v = act_in[rowbase + ix];
            Asf[ml >> 4][ahi * 16 + (ml & 15)][ae] = v;
        }
        asm volatile("s_wait_asynccnt 0x0" ::: "memory");
        __syncthreads();

        // ---- load fragments (one 32B LDS read each), 2*NT WMMAs ----
        v16bf a[2], b[NT];
#pragma unroll
        for (int i = 0; i < 2; ++i)
            a[i] = *(const v16bf*)(&Asf[mg * 2 + i][lane][0]);
#pragma unroll
        for (int j = 0; j < NT; ++j)
            b[j] = *(const v16bf*)(&Bsf[ng * NT + j][lane][0]);
#pragma unroll
        for (int i = 0; i < 2; ++i)
#pragma unroll
            for (int j = 0; j < NT; ++j)
                acc[i][j] = __builtin_amdgcn_wmma_f32_16x16x32_bf16(
                    false, a[i], false, b[j], (short)0, acc[i][j], false, false);
        __syncthreads();
    }

    // ---- epilogue: C/D layout -> acc[g] is M = mbase + g (contiguous) ----
#pragma unroll
    for (int j = 0; j < NT; ++j) {
        int ngl = n0 + (ng * NT + j) * 16 + l16;
        float bv = bias[ngl];
#pragma unroll
        for (int i = 0; i < 2; ++i) {
            int mbase = m0 + (mg * 2 + i) * 16 + 8 * hi;
            float r[8];
#pragma unroll
            for (int g = 0; g < 8; ++g) r[g] = acc[i][j][g] + bv;
            if (ACT == 1) {
                uint4 p;
                unsigned int w0[4];
#pragma unroll
                for (int q = 0; q < 4; ++q) {
                    float lo = r[2 * q]     > 0.f ? r[2 * q]     : 0.f;
                    float hh = r[2 * q + 1] > 0.f ? r[2 * q + 1] : 0.f;
                    w0[q] = (unsigned int)f2bf(lo) | ((unsigned int)f2bf(hh) << 16);
                }
                p.x = w0[0]; p.y = w0[1]; p.z = w0[2]; p.w = w0[3];
                *(uint4*)(&out_bf[ngl * HW_LR + mbase]) = p;
            } else {
                if (ACT == 2) {
#pragma unroll
                    for (int g = 0; g < 8; ++g) r[g] = 1.f / (1.f + expf(-r[g]));
                }
                float4 lo = {r[0], r[1], r[2], r[3]};
                float4 hh = {r[4], r[5], r[6], r[7]};
                *(float4*)(&out_f32[ngl * HW_LR + mbase])     = lo;
                *(float4*)(&out_f32[ngl * HW_LR + mbase + 4]) = hh;
            }
        }
    }
}

// ---------------------------------------------------------------------------
// Final fusion: pixel-shuffle combine of the two nets' kernels/offsets,
// mean-subtracted weights, deformable bilinear sampling of the 15x15 patch,
// residual add of x_up.  One thread per HR pixel.
// ---------------------------------------------------------------------------
__device__ __forceinline__ float patch_tap(const float* __restrict__ xup,
                                           int y, int x, int xi, int yi) {
    if (xi < 0 || xi >= 15 || yi < 0 || yi >= 15) return 0.f;
    int gy = y + yi - 7, gx = x + xi - 7;
    if (gy < 0 || gy >= W_HR || gx < 0 || gx >= W_HR) return 0.f;
    return xup[gy * W_HR + gx];
}

__global__ __launch_bounds__(256) void fuse_final(
    const float* __restrict__ imk, const float* __restrict__ imo,
    const float* __restrict__ dpk, const float* __restrict__ dpo,
    const float* __restrict__ xup, float* __restrict__ out) {
    int idx = blockIdx.x * 256 + threadIdx.x;
    if (idx >= HW_HR) return;
    int y = idx >> 9, x = idx & 511;
    int y4 = y >> 2, x4 = x >> 2;
    int ij = (y & 3) * 4 + (x & 3);
    int lr = y4 * W_LR + x4;

    float wgt[9];
    float mean = 0.f;
#pragma unroll
    for (int cc = 0; cc < 9; ++cc) {
        int ch = cc * 16 + ij;                       // pixel_shuffle channel map
        float wv = imk[ch * HW_LR + lr] * dpk[ch * HW_LR + lr];
        wgt[cc] = wv;
        mean += wv;
    }
    mean *= (1.f / 9.f);

    float acc = 0.f;
#pragma unroll
    for (int a = 0; a < 3; ++a) {
#pragma unroll
        for (int c = 0; c < 3; ++c) {
            int cc = a * 3 + c;
            int chx = ((cc * 2 + 0) * 16 + ij) * HW_LR + lr;
            int chy = ((cc * 2 + 1) * 16 + ij) * HW_LR + lr;
            float offx = imo[chx] * dpo[chx];
            float offy = imo[chy] * dpo[chy];
            // grid vals = [7,8,9]; normalize then align_corners=False unmap
            float cx = ((7.f + (float)a + offx) / 15.f) * 2.f - 1.f;
            float cy = ((7.f + (float)c + offy) / 15.f) * 2.f - 1.f;
            float ixf = ((cx + 1.f) * 15.f - 1.f) * 0.5f;
            float iyf = ((cy + 1.f) * 15.f - 1.f) * 0.5f;
            float x0f = floorf(ixf), y0f = floorf(iyf);
            float wx1 = ixf - x0f, wx0 = 1.f - wx1;
            float wy1 = iyf - y0f, wy0 = 1.f - wy1;
            int x0i = (int)x0f, y0i = (int)y0f;
            float s = wy0 * wx0 * patch_tap(xup, y, x, x0i,     y0i)
                    + wy0 * wx1 * patch_tap(xup, y, x, x0i + 1, y0i)
                    + wy1 * wx0 * patch_tap(xup, y, x, x0i,     y0i + 1)
                    + wy1 * wx1 * patch_tap(xup, y, x, x0i + 1, y0i + 1);
            acc += s * (wgt[cc] - mean);
        }
    }
    out[idx] = acc + xup[idx];
}

// ---------------------------------------------------------------------------
// Host launcher
// ---------------------------------------------------------------------------
extern "C" void kernel_launch(void* const* d_in, const int* in_sizes, int n_in,
                              void* d_out, int out_size, void* d_ws, size_t ws_size,
                              hipStream_t stream) {
    (void)in_sizes; (void)n_in; (void)out_size; (void)ws_size;

    char* ws = (char*)d_ws;
    size_t off = 0;
    auto alloc = [&](size_t bytes) -> char* {
        char* p = ws + off;
        off = (off + bytes + 255) & ~(size_t)255;
        return p;
    };

    float*          xup   = (float*)alloc(HW_HR * sizeof(float));
    unsigned short* re_im = (unsigned short*)alloc((size_t)160 * HW_LR * 2);
    unsigned short* re_dp = (unsigned short*)alloc((size_t)16 * HW_LR * 2);
    unsigned short* actA  = (unsigned short*)alloc((size_t)128 * HW_LR * 2);
    unsigned short* actB  = (unsigned short*)alloc((size_t)128 * HW_LR * 2);

    // per-layer geometry: K, real COUT, padded COUT, has-BN, param slot offset
    const int Kim[8] = {1440, 288, 288, 576, 576, 1152, 128, 128};
    const int Kdp[8] = { 144, 288, 288, 576, 576, 1152, 128, 128};
    const int Cr [8] = {  32,  32,  64,  64, 128,  128, 144, 288};
    const int Cp [8] = {  32,  32,  64,  64, 128,  128, 160, 288};
    const int hasBN[8] = {1, 0, 1, 0, 1, 0, 0, 0};
    const int pW[8] = {0, 6, 8, 14, 16, 22, 24, 26};   // weight slot within net

    unsigned short* wgt[2][8];
    float*          bia[2][8];
    for (int net = 0; net < 2; ++net)
        for (int l = 0; l < 8; ++l) {
            int K  = net ? Kdp[l] : Kim[l];
            int Kp = (K + 31) & ~31;
            wgt[net][l] = (unsigned short*)alloc((size_t)Kp * Cp[l] * 2);
            bia[net][l] = (float*)alloc((size_t)Cp[l] * sizeof(float));
        }

    float* imk = (float*)alloc((size_t)160 * HW_LR * sizeof(float));
    float* imo = (float*)alloc((size_t)288 * HW_LR * sizeof(float));
    float* dpk = (float*)alloc((size_t)160 * HW_LR * sizeof(float));
    float* dpo = (float*)alloc((size_t)288 * HW_LR * sizeof(float));

    const float* x_in = (const float*)d_in[0];
    const float* hrg  = (const float*)d_in[1];
    const int base[2] = {2, 30};   // image net leaves, then depth net leaves

    // 1) bicubic upsample
    bicubic_up<<<(HW_HR + 255) / 256, 256, 0, stream>>>(x_in, xup);
    // 2) space-to-depth (guidance uses hr input; depth uses x_up)
    s2d_bf16<<<(160 * HW_LR + 255) / 256, 256, 0, stream>>>(hrg, re_im, 10);
    s2d_bf16<<<(16 * HW_LR + 255) / 256, 256, 0, stream>>>(xup, re_dp, 1);

    // 3) weight prep (BN folded, bf16, fragment-ordered panels, padding)
    for (int net = 0; net < 2; ++net) {
        for (int l = 0; l < 8; ++l) {
            int K  = net ? Kdp[l] : Kim[l];
            int Kp = (K + 31) & ~31;
            const float* w = (const float*)d_in[base[net] + pW[l] + 0];
            const float* b = (const float*)d_in[base[net] + pW[l] + 1];
            const float *g = nullptr, *bb = nullptr, *m = nullptr, *v = nullptr;
            if (hasBN[l]) {
                g  = (const float*)d_in[base[net] + pW[l] + 2];
                bb = (const float*)d_in[base[net] + pW[l] + 3];
                m  = (const float*)d_in[base[net] + pW[l] + 4];
                v  = (const float*)d_in[base[net] + pW[l] + 5];
            }
            int total = Kp * Cp[l];
            prep_weights<<<(total + 255) / 256, 256, 0, stream>>>(
                w, b, g, bb, m, v, K, Kp, Cr[l], Cp[l], wgt[net][l], bia[net][l]);
        }
    }

    dim3 blk(256);
    // 4) image kernel-net  (grid.x = 16384/128 = 128, grid.y = COUTp/(32*NT))
    conv_wmma<160, 3,  32, 1, 1><<<dim3(128, 1), blk, 0, stream>>>(re_im, wgt[0][0], bia[0][0], actA, nullptr);
    conv_wmma< 32, 3,  32, 1, 1><<<dim3(128, 1), blk, 0, stream>>>(actA,  wgt[0][1], bia[0][1], actB, nullptr);
    conv_wmma< 32, 3,  64, 1, 2><<<dim3(128, 1), blk, 0, stream>>>(actB,  wgt[0][2], bia[0][2], actA, nullptr);
    conv_wmma< 64, 3,  64, 1, 2><<<dim3(128, 1), blk, 0, stream>>>(actA,  wgt[0][3], bia[0][3], actB, nullptr);
    conv_wmma< 64, 3, 128, 1, 2><<<dim3(128, 2), blk, 0, stream>>>(actB,  wgt[0][4], bia[0][4], actA, nullptr);
    conv_wmma<128, 3, 128, 1, 2><<<dim3(128, 2), blk, 0, stream>>>(actA,  wgt[0][5], bia[0][5], actB, nullptr);
    conv_wmma<128, 1, 160, 2, 1><<<dim3(128, 5), blk, 0, stream>>>(actB,  wgt[0][6], bia[0][6], nullptr, imk);
    conv_wmma<128, 1, 288, 0, 1><<<dim3(128, 9), blk, 0, stream>>>(actB,  wgt[0][7], bia[0][7], nullptr, imo);

    // 5) depth kernel-net
    conv_wmma< 16, 3,  32, 1, 1><<<dim3(128, 1), blk, 0, stream>>>(re_dp, wgt[1][0], bia[1][0], actA, nullptr);
    conv_wmma< 32, 3,  32, 1, 1><<<dim3(128, 1), blk, 0, stream>>>(actA,  wgt[1][1], bia[1][1], actB, nullptr);
    conv_wmma< 32, 3,  64, 1, 2><<<dim3(128, 1), blk, 0, stream>>>(actB,  wgt[1][2], bia[1][2], actA, nullptr);
    conv_wmma< 64, 3,  64, 1, 2><<<dim3(128, 1), blk, 0, stream>>>(actA,  wgt[1][3], bia[1][3], actB, nullptr);
    conv_wmma< 64, 3, 128, 1, 2><<<dim3(128, 2), blk, 0, stream>>>(actB,  wgt[1][4], bia[1][4], actA, nullptr);
    conv_wmma<128, 3, 128, 1, 2><<<dim3(128, 2), blk, 0, stream>>>(actA,  wgt[1][5], bia[1][5], actB, nullptr);
    conv_wmma<128, 1, 160, 2, 1><<<dim3(128, 5), blk, 0, stream>>>(actB,  wgt[1][6], bia[1][6], nullptr, dpk);
    conv_wmma<128, 1, 288, 0, 1><<<dim3(128, 9), blk, 0, stream>>>(actB,  wgt[1][7], bia[1][7], nullptr, dpo);

    // 6) deformable aggregation + residual
    fuse_final<<<(HW_HR + 255) / 256, 256, 0, stream>>>(imk, imo, dpk, dpo, xup, (float*)d_out);
}
